// VariationalGRU_26886495272995
// MI455X (gfx1250) — compile-verified, matrix operand order checked by reading, not run on hardware
//
#include <hip/hip_runtime.h>
#include <hip/hip_bf16.h>

// ---------------- problem constants ----------------
#define B_      1024
#define T_ENC_  128
#define T_DEC_  64
#define H_      128     // latent / mlp input width
#define F_      16
#define S_      8
#define H_ENC_  256     // encoder hidden per layer
#define G_ENC_  768     // 3*H_ENC_
#define H_DEC_  132     // decoder hidden per layer
#define LDH_    256     // LDS row stride (bf16 elems) for hidden buffers
#define BLOCK_  512     // 16 waves of 32

#define OFF_DEC_  ((long)S_ * B_ * (T_ENC_ - 1) * 2)   // dec_preds start in d_out
#define WS_LAST_  (2 * B_ * H_ENC_)                    // 'last' start in ws (floats)
#define WS_BF_    (WS_LAST_ + 2 * B_)                  // float offset of bf16 region

// packed-weight offsets (bf16 elements) inside the bf16 region.
// tile = 512 bf16 (32 lanes x 16), section-major: [sec][ntile][ktile][lane][16]
#define P_EWIH0_  0L                                    // 3 x 16 x 1 tiles
#define P_EWHH0_  (P_EWIH0_ + 3L * 16 * 1 * 512)        // 3 x 16 x 8
#define P_EWIH1_  (P_EWHH0_ + 3L * 16 * 8 * 512)
#define P_EWHH1_  (P_EWIH1_ + 3L * 16 * 8 * 512)
#define P_DWHH0_  (P_EWHH1_ + 3L * 16 * 8 * 512)        // 3 x 9 x 5
#define P_DWIH1_  (P_DWHH0_ + 3L * 9 * 5 * 512)
#define P_DWHH1_  (P_DWIH1_ + 3L * 9 * 5 * 512)
#define P_W1_     (P_DWHH1_ + 3L * 9 * 5 * 512)         // 1 x 4 x 4
#define P_END_    (P_W1_ + 1L * 4 * 4 * 512)

typedef __attribute__((ext_vector_type(16))) __bf16 v16bf;
typedef __attribute__((ext_vector_type(8)))  float  v8f;

// ---------------- math helpers ----------------
__device__ __forceinline__ float sigmoid_(float x) { return 1.0f / (1.0f + __expf(-x)); }
__device__ __forceinline__ float tanh_(float x) {
  float e = __expf(-2.0f * x);
  return (1.0f - e) / (1.0f + e);
}
__device__ __forceinline__ float softplus_(float x) {
  return (x > 20.0f) ? x : __logf(1.0f + __expf(x));
}

// ---------------- WMMA helpers ----------------
__device__ __forceinline__ v8f wmma_bf16(v16bf a, v16bf b, v8f c) {
  return __builtin_amdgcn_wmma_f32_16x16x32_bf16(false, a, false, b, (short)0, c,
                                                 false, false);
}

// A fragment (16x32 bf16) from LDS -> two ds_load_b128 per lane.
__device__ __forceinline__ v16bf load_a(const __bf16* __restrict__ sh, int ld,
                                        int k0, int lane) {
  const __bf16* rp = sh + (lane & 15) * ld + k0 + ((lane >> 4) << 3);
  v16bf a;
#pragma unroll
  for (int j = 0; j < 8; ++j) { a[j] = rp[j]; a[j + 8] = rp[16 + j]; }
  return a;
}

// Three gate tiles (r,z,n) for this wave's 16 hidden units from packed bf16
// weights.  Only the integer tile offset is made opaque (empty asm), so LICM
// cannot hoist the (time-invariant) weight loads out of the recurrence loop --
// no scratch spills -- while the pointer keeps its inferred GLOBAL address
// space (global_load_b128, LOADcnt only, no FLAT aperture checks).
// Accumulators are born at the first WMMA with inline zero C; biases are
// applied in the gating epilogue.
__device__ __forceinline__ void gemm3p(v8f (&acc)[3],
                                       const __bf16* __restrict__ shA, int lda,
                                       int ktiles,
                                       const __bf16* __restrict__ P, long secsz,
                                       int c0, int lane) {
  const long lbase = ((long)(c0 >> 4) * ktiles * 32 + lane) * 16;
  {
    long off = lbase;
    asm volatile("" : "+v"(off));
    v16bf a = load_a(shA, lda, 0, lane);
    v8f z = {0.f, 0.f, 0.f, 0.f, 0.f, 0.f, 0.f, 0.f};
#pragma unroll
    for (int g = 0; g < 3; ++g)
      acc[g] = wmma_bf16(a, *(const v16bf*)(P + off + g * secsz), z);
  }
  for (int kt = 1; kt < ktiles; ++kt) {
    long off = lbase + (long)kt * 512;
    asm volatile("" : "+v"(off));
    v16bf a = load_a(shA, lda, kt * 32, lane);
#pragma unroll
    for (int g = 0; g < 3; ++g)
      acc[g] = wmma_bf16(a, *(const v16bf*)(P + off + g * secsz), acc[g]);
  }
}

// Per-lane gate biases, computed once before the time loop:
// x = bih_r+bhh_r, y = bih_z+bhh_z, z = bih_n, w = bhh_n  (biha may be null).
__device__ __forceinline__ float4 bias4(const float* __restrict__ biha,
                                        const float* __restrict__ bhha,
                                        int bsec, int bmax, int c0, int lane) {
  int j = c0 + (lane & 15);
  int nr = j;            if (nr > bmax) nr = bmax;
  int nz = bsec + j;     if (nz > bmax) nz = bmax;
  int nn = 2 * bsec + j; if (nn > bmax) nn = bmax;
  float4 r;
  r.x = (biha ? biha[nr] : 0.0f) + bhha[nr];
  r.y = (biha ? biha[nz] : 0.0f) + bhha[nz];
  r.z = biha ? biha[nn] : 0.0f;
  r.w = bhha[nn];
  return r;
}

// In-register GRU gating (biases folded in); writes new h (and optionally the
// masked MLP input) back to LDS.
__device__ __forceinline__ void gate_update(const v8f (&gi)[3], const v8f (&gh)[3],
                                            float4 bs, __bf16* __restrict__ shh,
                                            int ldh, int c0, int hid,
                                            const int* __restrict__ mrow,
                                            __bf16* __restrict__ shmo, int molim,
                                            int lane) {
  int j = c0 + (lane & 15);
  bool valid = j < hid;
  int jc = valid ? j : 0;
#pragma unroll
  for (int r = 0; r < 8; ++r) {
    int M = r + ((lane >> 4) << 3);
    float hold = (float)shh[M * ldh + jc];
    float rg = sigmoid_(gi[0][r] + gh[0][r] + bs.x);
    float zg = sigmoid_(gi[1][r] + gh[1][r] + bs.y);
    float ng = tanh_((gi[2][r] + bs.z) + rg * (gh[2][r] + bs.w));
    float hn = (1.0f - zg) * ng + zg * hold;
    bool keep = mrow ? (mrow[M] != 0) : true;
    float ho = keep ? hn : hold;
    if (valid) {
      shh[M * ldh + j] = (__bf16)ho;
      if (shmo && j < molim) shmo[M * molim + j] = (__bf16)(keep ? ho : 0.0f);
    }
  }
}

// MLP layer 1: relu(in[16x128] @ W1.T + b1) via WMMA on packed W1 (waves 0..3).
// W1 fragments are intentionally hoistable (32 VGPRs, stays register-resident).
__device__ __forceinline__ void mlp_l1(const __bf16* __restrict__ shmo,
                                       float* __restrict__ shmlp,
                                       const __bf16* __restrict__ PW1,
                                       const float* __restrict__ b1,
                                       int wave, int lane) {
  if (wave < 4) {
    int nb = wave * 16;
    float bv = b1[nb + (lane & 15)];
    const __bf16* pw = PW1 + (((long)wave * 4) * 32 + lane) * 16;
    v8f acc;
    {
      v16bf a = load_a(shmo, H_, 0, lane);
      v8f z = {0.f, 0.f, 0.f, 0.f, 0.f, 0.f, 0.f, 0.f};
      acc = wmma_bf16(a, *(const v16bf*)pw, z);
    }
#pragma unroll
    for (int kt = 1; kt < 4; ++kt) {
      v16bf a = load_a(shmo, H_, kt * 32, lane);
      acc = wmma_bf16(a, *(const v16bf*)(pw + (long)kt * 512), acc);
    }
#pragma unroll
    for (int r = 0; r < 8; ++r) {
      int M = r + ((lane >> 4) << 3);
      float v = acc[r] + bv;
      shmlp[M * 64 + nb + (lane & 15)] = v > 0.0f ? v : 0.0f;
    }
  }
}

// MLP layer 2 + softplus: one lane per (row, out-channel) pair, wave 0 only.
__device__ __forceinline__ float mlp_l2(const float* __restrict__ shmlp,
                                        const float* __restrict__ W2,
                                        const float* __restrict__ b2, int lane) {
  int rr = lane >> 1, o = lane & 1;
  float a = b2[o];
#pragma unroll 8
  for (int jj = 0; jj < 64; ++jj) a += shmlp[rr * 64 + jj] * W2[o * 64 + jj];
  return softplus_(a);
}

// ======================= weight pack kernel =======================
__global__ void pack_w(const float* __restrict__ W, int ldw, int secStride,
                       int nvalid, int kvalid, int nsec, int ntiles, int ktiles,
                       __hip_bfloat16* __restrict__ dst_) {
  __bf16* __restrict__ dst = (__bf16*)dst_;
  long total = (long)nsec * ntiles * ktiles * 512;
  for (long idx = (long)blockIdx.x * blockDim.x + threadIdx.x; idx < total;
       idx += (long)gridDim.x * blockDim.x) {
    int j = (int)(idx & 15);
    int l = (int)((idx >> 4) & 31);
    long tile = idx >> 9;
    int kt = (int)(tile % ktiles);
    long nts = tile / ktiles;
    int nt = (int)(nts % ntiles);
    int sec = (int)(nts / ntiles);
    int n = nt * 16 + (l & 15);
    int k = kt * 32 + ((l >> 4) << 3) + (j < 8 ? j : 8 + j);
    float v = 0.0f;
    if (n < nvalid && k < kvalid)
      v = W[(long)(sec * secStride + n) * ldw + k];
    dst[idx] = (__bf16)v;
  }
}

// ======================= encoder kernel =======================
__global__ __launch_bounds__(BLOCK_) void vgru_encoder(
    const float* __restrict__ x, const float* __restrict__ feat,
    const int* __restrict__ mask,
    const float* __restrict__ bih0, const float* __restrict__ bhh0,
    const float* __restrict__ bih1, const float* __restrict__ bhh1,
    const float* __restrict__ b1,
    const float* __restrict__ W2, const float* __restrict__ b2,
    const __hip_bfloat16* __restrict__ wb_, float* __restrict__ out,
    float* __restrict__ ws) {
  const __bf16* __restrict__ wb = (const __bf16*)wb_;
  __shared__ __bf16 sh_h0[16 * LDH_];
  __shared__ __bf16 sh_h1[16 * LDH_];
  __shared__ __bf16 sh_in[16 * 32];
  __shared__ __bf16 sh_mo[16 * H_];
  __shared__ float  sh_mlp[16 * 64];
  __shared__ int    sh_m[16];

  const int tid = threadIdx.x;
  const int wave = tid >> 5, lane = tid & 31;
  const int b0 = blockIdx.x * 16;
  const int c0 = wave * 16;

  __builtin_prefetch(wb + P_EWHH0_, 0, 3);
  __builtin_prefetch(wb + P_EWIH1_, 0, 3);
  __builtin_prefetch(wb + P_EWHH1_, 0, 3);

  for (int i = tid; i < 16 * LDH_; i += BLOCK_) {
    sh_h0[i] = (__bf16)0.0f;
    sh_h1[i] = (__bf16)0.0f;
  }

  // per-lane gate biases (time-invariant)
  const float4 bs0 = bias4(bih0, bhh0, H_ENC_, G_ENC_ - 1, c0, lane);
  const float4 bs1 = bias4(bih1, bhh1, H_ENC_, G_ENC_ - 1, c0, lane);
  __syncthreads();

  const long SE1 = 16L * 1 * 512;
  const long SE8 = 16L * 8 * 512;

  for (int t = 0; t < T_ENC_; ++t) {
    for (int i = tid; i < 16 * 32; i += BLOCK_) {
      int rr = i >> 5, cc = i & 31;
      int b = b0 + rr;
      float v = 0.0f;
      if (cc < 2)       v = x[((long)b * T_ENC_ + t) * 2 + cc];
      else if (cc < 18) v = feat[((long)b * (T_ENC_ + T_DEC_) + t) * F_ + (cc - 2)];
      sh_in[i] = (__bf16)v;
    }
    if (tid < 16) sh_m[tid] = mask[(b0 + tid) * T_ENC_ + t];
    __syncthreads();

    v8f gi[3], gh[3];
    gemm3p(gi, sh_in, 32, 1, wb + P_EWIH0_, SE1, c0, lane);
    gemm3p(gh, sh_h0, LDH_, 8, wb + P_EWHH0_, SE8, c0, lane);
    __syncthreads();
    gate_update(gi, gh, bs0, sh_h0, LDH_, c0, H_ENC_, sh_m, nullptr, 0, lane);
    __syncthreads();

    gemm3p(gi, sh_h0, LDH_, 8, wb + P_EWIH1_, SE8, c0, lane);
    gemm3p(gh, sh_h1, LDH_, 8, wb + P_EWHH1_, SE8, c0, lane);
    __syncthreads();
    gate_update(gi, gh, bs1, sh_h1, LDH_, c0, H_ENC_, sh_m, sh_mo, H_, lane);
    __syncthreads();

    mlp_l1(sh_mo, sh_mlp, wb + P_W1_, b1, wave, lane);
    __syncthreads();
    if (wave == 0) {
      float p = mlp_l2(sh_mlp, W2, b2, lane);
      int rr = lane >> 1, o = lane & 1;
      int b = b0 + rr;
      if (t < T_ENC_ - 1) {
#pragma unroll
        for (int s = 0; s < S_; ++s)
          out[(((long)s * B_ + b) * (T_ENC_ - 1) + t) * 2 + o] = p;
      } else {
        ws[WS_LAST_ + b * 2 + o] = p;
      }
    }
    __syncthreads();
  }

  for (int i = tid; i < 16 * H_ENC_; i += BLOCK_) {
    int rr = i / H_ENC_, cc = i % H_ENC_;
    int b = b0 + rr;
    ws[((long)0 * B_ + b) * H_ENC_ + cc] = (float)sh_h0[rr * LDH_ + cc];
    ws[((long)1 * B_ + b) * H_ENC_ + cc] = (float)sh_h1[rr * LDH_ + cc];
  }
}

// ======================= decoder kernel =======================
__global__ __launch_bounds__(BLOCK_) void vgru_decoder(
    const float* __restrict__ eps, const float* __restrict__ feat,
    const float* __restrict__ Wih0, const float* __restrict__ bih0,
    const float* __restrict__ bhh0, const float* __restrict__ bih1,
    const float* __restrict__ bhh1, const float* __restrict__ b1,
    const float* __restrict__ W2, const float* __restrict__ b2,
    const __hip_bfloat16* __restrict__ wb_, float* __restrict__ out,
    const float* __restrict__ ws) {
  const __bf16* __restrict__ wb = (const __bf16*)wb_;
  __shared__ __bf16 sh_d0[16 * LDH_];
  __shared__ __bf16 sh_d1[16 * LDH_];
  __shared__ __bf16 sh_mo[16 * H_];
  __shared__ float  sh_mlp[16 * 64];
  __shared__ float  sh_po[16 * 2];

  const int tid = threadIdx.x;
  const int wave = tid >> 5, lane = tid & 31;
  const int s = blockIdx.x >> 6;
  const int b0 = (blockIdx.x & 63) * 16;
  const int c0 = wave * 16;
  const bool wvalid = (c0 < 144);

  __builtin_prefetch(wb + P_DWHH0_, 0, 3);
  __builtin_prefetch(wb + P_DWIH1_, 0, 3);
  __builtin_prefetch(wb + P_DWHH1_, 0, 3);

  for (int i = tid; i < 16 * LDH_; i += BLOCK_) {
    int rr = i / LDH_, cc = i % LDH_;
    int b = b0 + rr;
    float v0 = 0.0f, v1 = 0.0f;
    if (cc < H_) {
      float mu0 = ws[((long)0 * B_ + b) * H_ENC_ + cc];
      float lv0 = ws[((long)0 * B_ + b) * H_ENC_ + H_ + cc];
      float mu1 = ws[((long)1 * B_ + b) * H_ENC_ + cc];
      float lv1 = ws[((long)1 * B_ + b) * H_ENC_ + H_ + cc];
      v0 = eps[(((long)s * 2 + 0) * B_ + b) * H_ + cc] * __expf(0.5f * lv0) + mu0;
      v1 = eps[(((long)s * 2 + 1) * B_ + b) * H_ + cc] * __expf(0.5f * lv1) + mu1;
    } else if (cc < H_ + 4) {
      float fc = feat[((long)b * (T_ENC_ + T_DEC_) + T_ENC_) * F_ + (cc - H_)];
      v0 = fc;
      v1 = fc;
    }
    sh_d0[i] = (__bf16)v0;
    sh_d1[i] = (__bf16)v1;
  }
  if (tid < 32) {
    int rr = tid >> 1, o = tid & 1;
    float v = ws[WS_LAST_ + (b0 + rr) * 2 + o];
    sh_po[rr * 2 + o] = v;
    out[OFF_DEC_ + (((long)s * B_ + (b0 + rr)) * T_DEC_ + 0) * 2 + o] = v;
  }

  // time-invariant per-lane constants
  const float4 bsd0 = bias4(nullptr, bhh0, H_DEC_, 3 * H_DEC_ - 1, c0, lane);
  const float4 bsd1 = bias4(bih1, bhh1, H_DEC_, 3 * H_DEC_ - 1, c0, lane);
  float w0g[3], w1g[3], bbg[3];
  {
    int jr = c0 + (lane & 15);
    int j = jr < H_DEC_ ? jr : H_DEC_ - 1;
#pragma unroll
    for (int g = 0; g < 3; ++g) {
      int n = g * H_DEC_ + j;
      w0g[g] = Wih0[n * 2 + 0];
      w1g[g] = Wih0[n * 2 + 1];
      bbg[g] = bih0[n];
    }
  }
  __syncthreads();

  const long SD = 9L * 5 * 512;

  for (int t = 1; t < T_DEC_; ++t) {
    v8f gi[3], gh[3];

    if (wvalid) {
      // layer 0: gi (K=2) on VALU from register-resident weights
#pragma unroll
      for (int r = 0; r < 8; ++r) {
        int M = r + ((lane >> 4) << 3);
        float p0 = sh_po[M * 2 + 0], p1 = sh_po[M * 2 + 1];
#pragma unroll
        for (int g = 0; g < 3; ++g)
          gi[g][r] = bbg[g] + w0g[g] * p0 + w1g[g] * p1;
      }
      gemm3p(gh, sh_d0, LDH_, 5, wb + P_DWHH0_, SD, c0, lane);
    }
    __syncthreads();
    if (wvalid)
      gate_update(gi, gh, bsd0, sh_d0, LDH_, c0, H_DEC_, nullptr, nullptr, 0, lane);
    __syncthreads();

    if (wvalid) {
      gemm3p(gi, sh_d0, LDH_, 5, wb + P_DWIH1_, SD, c0, lane);
      gemm3p(gh, sh_d1, LDH_, 5, wb + P_DWHH1_, SD, c0, lane);
    }
    __syncthreads();
    if (wvalid)
      gate_update(gi, gh, bsd1, sh_d1, LDH_, c0, H_DEC_, nullptr, sh_mo, H_, lane);
    __syncthreads();

    mlp_l1(sh_mo, sh_mlp, wb + P_W1_, b1, wave, lane);
    __syncthreads();
    if (wave == 0) {
      float p = mlp_l2(sh_mlp, W2, b2, lane);
      int rr = lane >> 1, o = lane & 1;
      sh_po[rr * 2 + o] = p;
      out[OFF_DEC_ + (((long)s * B_ + (b0 + rr)) * T_DEC_ + t) * 2 + o] = p;
    }
    __syncthreads();
  }
}

// ======================= host launch =======================
extern "C" void kernel_launch(void* const* d_in, const int* in_sizes, int n_in,
                              void* d_out, int out_size, void* d_ws, size_t ws_size,
                              hipStream_t stream) {
  (void)in_sizes; (void)n_in; (void)out_size; (void)ws_size;
  const float* x     = (const float*)d_in[0];
  const float* feat  = (const float*)d_in[1];
  const int*   mask  = (const int*)d_in[2];
  const float* eps   = (const float*)d_in[3];
  const float* eWih0 = (const float*)d_in[7];
  const float* eWhh0 = (const float*)d_in[8];
  const float* ebih0 = (const float*)d_in[9];
  const float* ebhh0 = (const float*)d_in[10];
  const float* eWih1 = (const float*)d_in[11];
  const float* eWhh1 = (const float*)d_in[12];
  const float* ebih1 = (const float*)d_in[13];
  const float* ebhh1 = (const float*)d_in[14];
  const float* dWih0 = (const float*)d_in[15];
  const float* dWhh0 = (const float*)d_in[16];
  const float* dbih0 = (const float*)d_in[17];
  const float* dbhh0 = (const float*)d_in[18];
  const float* dWih1 = (const float*)d_in[19];
  const float* dWhh1 = (const float*)d_in[20];
  const float* dbih1 = (const float*)d_in[21];
  const float* dbhh1 = (const float*)d_in[22];
  const float* W1    = (const float*)d_in[23];
  const float* b1    = (const float*)d_in[24];
  const float* W2    = (const float*)d_in[25];
  const float* b2    = (const float*)d_in[26];
  float* out = (float*)d_out;
  float* ws  = (float*)d_ws;
  __hip_bfloat16* wb = (__hip_bfloat16*)(ws + WS_BF_);

  pack_w<<<32, 256, 0, stream>>>(eWih0, 18, H_ENC_, H_ENC_, 18, 3, 16, 1,
                                 wb + P_EWIH0_);
  pack_w<<<128, 256, 0, stream>>>(eWhh0, H_ENC_, H_ENC_, H_ENC_, H_ENC_, 3, 16, 8,
                                  wb + P_EWHH0_);
  pack_w<<<128, 256, 0, stream>>>(eWih1, H_ENC_, H_ENC_, H_ENC_, H_ENC_, 3, 16, 8,
                                  wb + P_EWIH1_);
  pack_w<<<128, 256, 0, stream>>>(eWhh1, H_ENC_, H_ENC_, H_ENC_, H_ENC_, 3, 16, 8,
                                  wb + P_EWHH1_);
  pack_w<<<64, 256, 0, stream>>>(dWhh0, H_DEC_, H_DEC_, H_DEC_, H_DEC_, 3, 9, 5,
                                 wb + P_DWHH0_);
  pack_w<<<64, 256, 0, stream>>>(dWih1, H_DEC_, H_DEC_, H_DEC_, H_DEC_, 3, 9, 5,
                                 wb + P_DWIH1_);
  pack_w<<<64, 256, 0, stream>>>(dWhh1, H_DEC_, H_DEC_, H_DEC_, H_DEC_, 3, 9, 5,
                                 wb + P_DWHH1_);
  pack_w<<<16, 256, 0, stream>>>(W1, H_, 64, 64, H_, 1, 4, 4, wb + P_W1_);

  vgru_encoder<<<B_ / 16, BLOCK_, 0, stream>>>(
      x, feat, mask, ebih0, ebhh0, ebih1, ebhh1, b1, W2, b2, wb, out, ws);

  vgru_decoder<<<S_ * (B_ / 16), BLOCK_, 0, stream>>>(
      eps, feat, dWih0, dbih0, dbhh0, dbih1, dbhh1, b1, W2, b2, wb, out, ws);
}
